// FlashAttention_11776800325841
// MI455X (gfx1250) — matrix operand order repaired and checked by reading, not compiled, not run on hardware
//
#include <hip/hip_runtime.h>
#include <math.h>

// MI455X (gfx1250) flash attention forward, wave32 + v_wmma_f32_16x16x32_bf16.
// B=4 H=12 S=4096 D=64, fp32 in/out, bf16 matmuls with fp32 accumulation.
// Transposed-score formulation: S^T = K*Q^T, O^T = V^T*P^T.
//  - lane = query: softmax max is an in-lane v_max3 tree + one v_permlanex16
//  - P^T B-fragments built in registers (permlanex16 half-swap), no LDS round-trip
//  - row-sum via ones-A WMMA; softmax in log2 space (bare v_exp_f32)
//  - 18 WMMA per 64-key wave-iteration

typedef __bf16 bf16_t;
typedef __attribute__((ext_vector_type(16))) __bf16 v16bf;
typedef __attribute__((ext_vector_type(8)))  float  v8f;

#define B_    4
#define H_    12
#define S_    4096
#define D_    64
#define QTILE 128   // query rows per block: 8 waves x 16 rows
#define KSTEP 64    // keys per inner iteration

#define WMMA_BF16(A, B, C) \
  __builtin_amdgcn_wmma_f32_16x16x32_bf16(false, (A), false, (B), (short)0, (C), false, false)

// swap data between the two 16-lane halves of a wave32 (identity lane select)
__device__ __forceinline__ unsigned permx16_u(unsigned x) {
  return (unsigned)__builtin_amdgcn_permlanex16((int)x, (int)x,
                                                0x76543210, 0xfedcba98, false, false);
}
__device__ __forceinline__ float permx16_f(float x) {
  return __int_as_float((int)permx16_u((unsigned)__float_as_int(x)));
}

__device__ __forceinline__ unsigned pk_bf16(float a, float b) {
  union { bf16_t h[2]; unsigned u; } t;
  t.h[0] = (bf16_t)a; t.h[1] = (bf16_t)b;
  return t.u;
}

union FragB { v16bf f; unsigned u[8]; };

__global__ __launch_bounds__(256)
void flash_fwd_gfx1250(const float* __restrict__ q,
                       const float* __restrict__ k,
                       const float* __restrict__ v,
                       float* __restrict__ out) {
  __shared__ bf16_t Kt[KSTEP][D_ + 8];        // [key][dim]
  __shared__ bf16_t Vt[D_][KSTEP + 8];        // [dim][key]  (transposed)

  const int tid  = threadIdx.x;
  const int wave = tid >> 5;
  const int lane = tid & 31;
  const int hlf  = lane >> 4;
  const int l16  = lane & 15;

  const int    bh   = blockIdx.y;                     // fused b*H+h, 0..47
  const size_t head = (size_t)bh * S_ * D_;
  const int   qbase = blockIdx.x * QTILE + wave * 16;

  // scale*log2(e) folded into Q: scores emerge in log2 space, ready for exp2
  const float qscale = 0.125f * 1.44269504088896340736f;

  // ---- Q^T B-fragments (32x16 bf16): lane = query col, elems = dims ----
  v16bf bq0, bq1;
  {
    const float* qp = q + head + (size_t)(qbase + l16) * D_;
    #pragma unroll
    for (int e = 0; e < 16; ++e) {
      bq0[e] = (bf16_t)(qp[     hlf * 16 + e] * qscale);
      bq1[e] = (bf16_t)(qp[32 + hlf * 16 + e] * qscale);
    }
  }

  // all-ones A fragment: 1^T * P^T gives the softmax denominator per query
  v16bf aones;
  #pragma unroll
  for (int e = 0; e < 16; ++e) aones[e] = (bf16_t)1.0f;

  v8f acc[4] = {{}, {}, {}, {}};   // O^T tiles: lane = query, rows = dims
  v8f lacc   = {};                 // row-sum (replicated across rows)
  float mrow = -INFINITY;          // per-lane = per-query running max (log2)

  for (int kb = 0; kb < S_; kb += KSTEP) {
    const int key = tid >> 2;            // 0..63
    const int d0  = (tid & 3) << 4;      // 0,16,32,48

    // prefetch next tile while we work on this one (L2 is 192MB; keep it hot)
    if (kb + KSTEP < S_) {
      __builtin_prefetch(k + head + (size_t)(kb + KSTEP + key) * D_ + d0, 0, 0);
      __builtin_prefetch(v + head + (size_t)(kb + KSTEP + key) * D_ + d0, 0, 0);
    }

    // ---- stage K/V tile (fp32 -> bf16), V transposed; fully coalesced ----
    {
      const float* kp = k + head + (size_t)(kb + key) * D_ + d0;
      const float* vp = v + head + (size_t)(kb + key) * D_ + d0;
      #pragma unroll
      for (int i = 0; i < 16; ++i) {
        Kt[key][d0 + i] = (bf16_t)kp[i];
        Vt[d0 + i][key] = (bf16_t)vp[i];
      }
    }
    __syncthreads();

    // ---- S^T = K * Q^T: four 16(key)x16(query) tiles via 8 WMMA ----
    // st[t][r] = score(query=l16, key = t*16 + r + 8*hlf), log2 space
    v8f st[4];
    #pragma unroll
    for (int t = 0; t < 4; ++t) {
      v16bf a0, a1;  // A 16x32: lane = key row, elems = dims (runs of 8)
      #pragma unroll
      for (int e = 0; e < 8; ++e) {
        a0[e]     = Kt[t * 16 + l16][     hlf * 16 + e];
        a0[e + 8] = Kt[t * 16 + l16][ 8 + hlf * 16 + e];
        a1[e]     = Kt[t * 16 + l16][32 + hlf * 16 + e];
        a1[e + 8] = Kt[t * 16 + l16][40 + hlf * 16 + e];
      }
      v8f s = {};
      s = WMMA_BF16(a0, bq0, s);
      s = WMMA_BF16(a1, bq1, s);
      st[t] = s;
    }

    // ---- online softmax: in-lane max tree + one cross-half exchange ----
    float lm = st[0][0];
    #pragma unroll
    for (int t = 0; t < 4; ++t)
      #pragma unroll
      for (int r = 0; r < 8; ++r) lm = fmaxf(lm, st[t][r]);
    const float cm = fmaxf(lm, permx16_f(lm));       // max over all 64 keys
    const float mn = fmaxf(mrow, cm);
    const float al = __builtin_amdgcn_exp2f(mrow - mn);   // old_scale
    const float off = mn + (mn - cm);  // exp2(s-off) == cur_exp*cur_scale (ref)
    mrow = mn;
    #pragma unroll
    for (int r = 0; r < 8; ++r) {
      acc[0][r] *= al; acc[1][r] *= al; acc[2][r] *= al; acc[3][r] *= al;
      lacc[r]   *= al;
    }

    // ---- P = exp2(st - off), still in S^T register layout ----
    #pragma unroll
    for (int t = 0; t < 4; ++t)
      #pragma unroll
      for (int r = 0; r < 8; ++r)
        st[t][r] = __builtin_amdgcn_exp2f(st[t][r] - off);

    // ---- build P^T B-fragments (32 keys x 16 queries) per K-slice and
    //      accumulate O^T += V^T * P^T (+ ones row-sum) : 10 WMMA ----
    #pragma unroll
    for (int s = 0; s < 2; ++s) {
      const int t0 = 2 * s, t1 = 2 * s + 1;
      unsigned keep[4], send[4];
      #pragma unroll
      for (int j = 0; j < 4; ++j) {
        const unsigned pk0 = pk_bf16(st[t0][2 * j], st[t0][2 * j + 1]);
        const unsigned pk1 = pk_bf16(st[t1][2 * j], st[t1][2 * j + 1]);
        keep[j] = hlf ? pk1 : pk0;   // the 8 keys this lane contributes in-place
        send[j] = hlf ? pk0 : pk1;   // the 8 keys the other half needs
      }
      FragB bp;
      #pragma unroll
      for (int j = 0; j < 4; ++j) {
        const unsigned recv = permx16_u(send[j]);
        bp.u[j]     = hlf ? recv    : keep[j];
        bp.u[4 + j] = hlf ? keep[j] : recv;
      }
      #pragma unroll
      for (int t = 0; t < 4; ++t) {
        v16bf av;  // A = V^T 16(dim)x32(key): lane = dim row, elems = keys
        #pragma unroll
        for (int e = 0; e < 8; ++e) {
          av[e]     = Vt[t * 16 + l16][s * 32 +     hlf * 16 + e];
          av[e + 8] = Vt[t * 16 + l16][s * 32 + 8 + hlf * 16 + e];
        }
        acc[t] = WMMA_BF16(av, bp.f, acc[t]);
      }
      lacc = WMMA_BF16(aones, bp.f, lacc);
    }

    __syncthreads();  // protect Kt/Vt before next staging round
  }

  // ---- finalize: out = acc / l; O^T layout -> 8 contiguous dims per lane ----
  const float inv = 1.0f / lacc[0];   // all rows of lacc are identical
  float* op = out + head + (size_t)(qbase + l16) * D_ + 8 * hlf;
  #pragma unroll
  for (int t = 0; t < 4; ++t)
    #pragma unroll
    for (int r = 0; r < 8; ++r)
      op[t * 16 + r] = acc[t][r] * inv;   // merges into global_store_b128

  (void)wave;
}

extern "C" void kernel_launch(void* const* d_in, const int* /*in_sizes*/, int /*n_in*/,
                              void* d_out, int /*out_size*/, void* /*d_ws*/, size_t /*ws_size*/,
                              hipStream_t stream) {
  const float* q = (const float*)d_in[0];
  const float* k = (const float*)d_in[1];
  const float* v = (const float*)d_in[2];
  dim3 grid(S_ / QTILE, B_ * H_);     // (32, 48)
  flash_fwd_gfx1250<<<grid, 256, 0, stream>>>(q, k, v, (float*)d_out);
}
